// CustomMultiheadAttention_53455162966536
// MI455X (gfx1250) — compile-verified, matrix operand order checked by reading, not run on hardware
//
#include <hip/hip_runtime.h>
#include <hip/hip_bf16.h>
#include <math.h>

#define B_  4
#define T_  4096
#define S_  4096
#define D_  512
#define H_  8
#define HD_ 64
#define SCH 64           // S-chunk width in the flash loop

typedef __attribute__((ext_vector_type(16))) __bf16          v16bf;
typedef __attribute__((ext_vector_type(8)))  __bf16          v8bf;
typedef __attribute__((ext_vector_type(8)))  float           v8f;
typedef __attribute__((ext_vector_type(8)))  unsigned short  us8;
typedef __attribute__((ext_vector_type(4)))  float           f32x4;
typedef __attribute__((ext_vector_type(4)))  int             i32x4;

__device__ __forceinline__ unsigned short f2bf(float f) {
  unsigned int u = __float_as_uint(f);
  u += 0x7FFFu + ((u >> 16) & 1u);          // round-to-nearest-even
  return (unsigned short)(u >> 16);
}

// Load a 16x32 (rows x K) bf16 WMMA fragment from row-major LDS.
// Per CDNA5 16-bit A/B layout: lanes 0-15 hold K={0..7,16..23},
// lanes 16-31 hold K={8..15,24..31} -> two contiguous ds_load_b128 per lane.
__device__ __forceinline__ v16bf load_frag(const unsigned short* rowptr, int lane) {
  const int kb = (lane >> 4) << 3;          // 0 or 8
  v8bf lo = *(const v8bf*)(rowptr + kb);
  v8bf hi = *(const v8bf*)(rowptr + kb + 16);
  return __builtin_shufflevector(lo, hi, 0,1,2,3,4,5,6,7,8,9,10,11,12,13,14,15);
}

__device__ __forceinline__ v8f wmma_bf16(v16bf a, v16bf b, v8f c) {
  return __builtin_amdgcn_wmma_f32_16x16x32_bf16(false, a, false, b, (short)0, c,
                                                 false, false);
}

// ---- cross-lane max over 16-lane halves: DPP xor-butterfly (no DS waits) ----
#if __has_builtin(__builtin_amdgcn_update_dpp)
#define XMAX_STEP(x, ctrl)                                                     \
  do {                                                                         \
    int _t = __builtin_amdgcn_update_dpp(0, __float_as_int(x), (ctrl), 0xf,    \
                                         0xf, true);                           \
    (x) = fmaxf((x), __int_as_float(_t));                                      \
  } while (0)
__device__ __forceinline__ float rowmax16(float x) {
  XMAX_STEP(x, 0xB1);   // quad_perm [1,0,3,2]  (xor 1)
  XMAX_STEP(x, 0x4E);   // quad_perm [2,3,0,1]  (xor 2)
  XMAX_STEP(x, 0x141);  // row_half_mirror      (xor 7)
  XMAX_STEP(x, 0x140);  // row_mirror           (xor 15)
  return x;             // masks {1,2,7,15} span lanes 0..15 -> allreduce
}
#else
__device__ __forceinline__ float rowmax16(float x) {
  for (int mk = 8; mk >= 1; mk >>= 1) x = fmaxf(x, __shfl_xor(x, mk, 32));
  return x;
}
#endif

// ---- async global->LDS staging (ASYNCcnt path), with sync fallback ----------
// clang's builtin takes typed pointers: (int4 addrspace(1)*, int4 addrspace(3)*,
// imm offset, imm cpol) per the round-2 diagnostic.
#if __has_builtin(__builtin_amdgcn_global_load_async_to_lds_b128)
#define HAVE_ASYNC 1
#define ASYNC_CP16(ldst, gsrc)                                                 \
  __builtin_amdgcn_global_load_async_to_lds_b128(                              \
      (__attribute__((address_space(1))) i32x4*)(gsrc),                        \
      (__attribute__((address_space(3))) i32x4*)(ldst), 0, 0)
#define ASYNC_WAIT() asm volatile("s_wait_asynccnt 0x0" ::: "memory")
#else
#define HAVE_ASYNC 0
#endif

// ---------------------------------------------------------------------------
// GEMM: Out[m,n] = sum_k X[m,k] * W[n,k] + bias[n]   (torch Linear semantics)
// Block: 256 threads (8 waves), tile 128(M) x 64(N), K stepped by 32.
// ---------------------------------------------------------------------------
template <bool OUT_BF16>
__global__ __launch_bounds__(256)
void gemm_bias(const float* __restrict__ X, const float* __restrict__ W,
               const float* __restrict__ bias, void* __restrict__ Out) {
  __shared__ unsigned short As[128 * 32];
  __shared__ unsigned short Bs[64 * 32];

  const int tid  = threadIdx.x;
  const int lane = tid & 31;
  const int wave = tid >> 5;
  const int m0   = blockIdx.x * 128;
  const int n0   = blockIdx.y * 64;
  const int nl   = lane & 15;

  v8f acc[4] = {};

  for (int k0 = 0; k0 < D_; k0 += 32) {
    { // stage A tile 128x32 (f32 -> bf16)
      const int row = tid >> 1, col = (tid & 1) * 16;
      const float* src = X + (size_t)(m0 + row) * D_ + k0 + col;
      us8 c0, c1;
      #pragma unroll
      for (int i = 0; i < 2; ++i) {
        f32x4 a = *(const f32x4*)(src + i * 4);
        f32x4 b = *(const f32x4*)(src + 8 + i * 4);
        c0[i*4+0]=f2bf(a.x); c0[i*4+1]=f2bf(a.y); c0[i*4+2]=f2bf(a.z); c0[i*4+3]=f2bf(a.w);
        c1[i*4+0]=f2bf(b.x); c1[i*4+1]=f2bf(b.y); c1[i*4+2]=f2bf(b.z); c1[i*4+3]=f2bf(b.w);
      }
      *(us8*)&As[row * 32 + col]     = c0;
      *(us8*)&As[row * 32 + col + 8] = c1;
    }
    { // stage B tile 64x32 from W (row n, cols k)
      const int row = tid >> 2, col = (tid & 3) * 8;
      const float* src = W + (size_t)(n0 + row) * D_ + k0 + col;
      us8 c0;
      #pragma unroll
      for (int i = 0; i < 2; ++i) {
        f32x4 a = *(const f32x4*)(src + i * 4);
        c0[i*4+0]=f2bf(a.x); c0[i*4+1]=f2bf(a.y); c0[i*4+2]=f2bf(a.z); c0[i*4+3]=f2bf(a.w);
      }
      *(us8*)&Bs[row * 32 + col] = c0;
    }
    __syncthreads();

    v16bf a = load_frag(&As[(wave * 16 + nl) * 32], lane);
    #pragma unroll
    for (int j = 0; j < 4; ++j) {
      v16bf b = load_frag(&Bs[(j * 16 + nl) * 32], lane);
      acc[j] = wmma_bf16(a, b, acc[j]);
    }
    __syncthreads();
  }

  const int mh = (lane >> 4) * 8;   // C layout: n = lane&15, m = r + mh
  #pragma unroll
  for (int j = 0; j < 4; ++j) {
    const int col = n0 + j * 16 + nl;
    const float bv = bias[col];
    #pragma unroll
    for (int r = 0; r < 8; ++r) {
      const size_t row = (size_t)(m0 + wave * 16 + mh + r);
      const float val  = acc[j][r] + bv;
      if constexpr (OUT_BF16)
        ((unsigned short*)Out)[row * D_ + col] = f2bf(val);
      else
        ((float*)Out)[row * D_ + col] = val;
    }
  }
}

// ---------------------------------------------------------------------------
// Flash attention: block = 4 waves, one (b, h, 64 query rows) per block.
// Each wave owns 16 rows. S streamed in 64-wide chunks; online softmax in
// log2 domain; denominator accumulated via an all-ones WMMA column (o[4]).
// ---------------------------------------------------------------------------
__global__ __launch_bounds__(128)
void attention(const unsigned short* __restrict__ Qp,
               const unsigned short* __restrict__ Kp,
               const unsigned short* __restrict__ Vp,
               const unsigned char* __restrict__ mask,
               float* __restrict__ AttnOut) {
  __shared__ unsigned short ldsQ[64 * 64];     // [t][d]
  __shared__ unsigned short ldsK[SCH * 64];    // [s][d]
  __shared__ unsigned short ldsVt[64 * SCH];   // [d][s] (transposed)
  __shared__ unsigned short ldsP[4][16 * SCH]; // per-wave P tile [t][s]

  const int tid  = threadIdx.x;
  const int lane = tid & 31;
  const int wave = tid >> 5;
  const int nl   = lane & 15;
  const int t0   = blockIdx.x * 64;
  const int h    = blockIdx.y;
  const int b    = blockIdx.z;

  const size_t qbase  = ((size_t)b * T_ + t0) * D_ + h * HD_;
  const size_t kvbase = ((size_t)b * S_) * D_ + h * HD_;
  const unsigned char* mrow = mask + (size_t)b * S_;

  // stage Q tile 64x64 (bf16)
  #pragma unroll
  for (int i = 0; i < 4; ++i) {
    const int c = tid * 4 + i;               // 512 chunks of 8 elems
    const int row = c >> 3, cc = (c & 7) * 8;
#if HAVE_ASYNC
    ASYNC_CP16(&ldsQ[row * 64 + cc], &Qp[qbase + (size_t)row * D_ + cc]);
#else
    *(us8*)&ldsQ[row * 64 + cc] = *(const us8*)&Qp[qbase + (size_t)row * D_ + cc];
#endif
  }
#if HAVE_ASYNC
  ASYNC_WAIT();
#endif
  __syncthreads();

  const v16bf qf0 = load_frag(&ldsQ[(wave * 16 + nl) * 64 + 0],  lane);
  const v16bf qf1 = load_frag(&ldsQ[(wave * 16 + nl) * 64 + 32], lane);

  v16bf ones;                                 // all-ones B matrix -> row sums
  #pragma unroll
  for (int i = 0; i < 16; ++i) ones[i] = (__bf16)1.0f;

  v8f o[5] = {};                              // o[0..3]=PV tiles, o[4]=sum(P)
  float m_i[8];
  #pragma unroll
  for (int r = 0; r < 8; ++r) m_i[r] = -1e30f;

  // scale folded with log2(e): softmax computed in exp2 domain
  const float kscale = 0.125f * 1.44269504088896340736f;

  for (int s0 = 0; s0 < S_; s0 += SCH) {
    // stage K chunk [SCH][64]
    #pragma unroll
    for (int i = 0; i < 4; ++i) {
      const int c = tid * 4 + i;             // 512 chunks
      const int row = c >> 3, cc = (c & 7) * 8;
#if HAVE_ASYNC
      ASYNC_CP16(&ldsK[row * 64 + cc],
                 &Kp[kvbase + (size_t)(s0 + row) * D_ + cc]);
#else
      *(us8*)&ldsK[row * 64 + cc] =
          *(const us8*)&Kp[kvbase + (size_t)(s0 + row) * D_ + cc];
#endif
    }
    // stage V transposed [d][s]
    {
      const int srow = tid >> 1, dc = (tid & 1) * 32;
      const unsigned short* src = &Vp[kvbase + (size_t)(s0 + srow) * D_ + dc];
      us8 vv[4];
      #pragma unroll
      for (int i = 0; i < 4; ++i) vv[i] = *(const us8*)(src + i * 8);
      #pragma unroll
      for (int i = 0; i < 4; ++i)
        #pragma unroll
        for (int e = 0; e < 8; ++e)
          ldsVt[(dc + i * 8 + e) * SCH + srow] = vv[i][e];
    }
    if (s0 + SCH < S_) {                     // global_prefetch_b8 next chunk
      __builtin_prefetch(&Vp[kvbase + (size_t)(s0 + SCH + (tid >> 1)) * D_], 0, 0);
    }
#if HAVE_ASYNC
    ASYNC_WAIT();
#endif
    __syncthreads();

    // scores: four 16x16 tiles over s, each accumulated over d=64
    v8f c[4];
    #pragma unroll
    for (int j = 0; j < 4; ++j) {
      v16bf kf = load_frag(&ldsK[(j * 16 + nl) * 64 + 0], lane);
      v8f t = {};
      t = wmma_bf16(qf0, kf, t);
      kf = load_frag(&ldsK[(j * 16 + nl) * 64 + 32], lane);
      c[j] = wmma_bf16(qf1, kf, t);
    }

    float msk[4];
    #pragma unroll
    for (int j = 0; j < 4; ++j)
      msk[j] = mrow[s0 + j * 16 + nl] ? -INFINITY : 0.f;

    // online max (log2 domain) + rescale factors
    float x[4][8], al[8];
    #pragma unroll
    for (int r = 0; r < 8; ++r) {
      #pragma unroll
      for (int j = 0; j < 4; ++j) x[j][r] = c[j][r] * kscale + msk[j];
      float rm = fmaxf(fmaxf(x[0][r], x[1][r]), fmaxf(x[2][r], x[3][r]));
      rm = rowmax16(rm);
      const float nm = fmaxf(m_i[r], rm);
      al[r]  = exp2f(m_i[r] - nm);
      m_i[r] = nm;
    }
    #pragma unroll
    for (int dt = 0; dt < 5; ++dt)
      #pragma unroll
      for (int r = 0; r < 8; ++r) o[dt][r] *= al[r];

    // P = exp2(x - m) -> LDS (C-layout regs -> [t][s]) -> A-fragments
    unsigned short* Pw = ldsP[wave];
    const int mh = (lane >> 4) * 8;
    #pragma unroll
    for (int r = 0; r < 8; ++r)
      #pragma unroll
      for (int j = 0; j < 4; ++j)
        Pw[(mh + r) * SCH + j * 16 + nl] = f2bf(exp2f(x[j][r] - m_i[r]));

    const v16bf pf0 = load_frag(&Pw[nl * SCH + 0],  lane);
    const v16bf pf1 = load_frag(&Pw[nl * SCH + 32], lane);
    #pragma unroll
    for (int dt = 0; dt < 4; ++dt) {
      v16bf vf = load_frag(&ldsVt[(dt * 16 + nl) * SCH + 0], lane);
      o[dt] = wmma_bf16(pf0, vf, o[dt]);
      vf = load_frag(&ldsVt[(dt * 16 + nl) * SCH + 32], lane);
      o[dt] = wmma_bf16(pf1, vf, o[dt]);
    }
    o[4] = wmma_bf16(pf0, ones, o[4]);       // denominator, all lanes equal
    o[4] = wmma_bf16(pf1, ones, o[4]);
    __syncthreads();
  }

  // epilogue: O / l  (guard l==0 like nan_to_num)
  const int mh = (lane >> 4) * 8;
  #pragma unroll
  for (int r = 0; r < 8; ++r) {
    const float l   = o[4][r];
    const float inv = (l > 0.f) ? 1.0f / l : 0.f;
    const size_t row = (size_t)b * T_ + t0 + wave * 16 + mh + r;
    #pragma unroll
    for (int dt = 0; dt < 4; ++dt)
      AttnOut[row * D_ + h * HD_ + dt * 16 + nl] = o[dt][r] * inv;
  }
}

// ---------------------------------------------------------------------------
extern "C" void kernel_launch(void* const* d_in, const int* in_sizes, int n_in,
                              void* d_out, int out_size, void* d_ws, size_t ws_size,
                              hipStream_t stream) {
  (void)in_sizes; (void)n_in; (void)out_size; (void)ws_size;
  const float* q  = (const float*)d_in[0];
  const float* k  = (const float*)d_in[1];
  const float* v  = (const float*)d_in[2];
  const unsigned char* mask = (const unsigned char*)d_in[3];
  const float* Wq = (const float*)d_in[4];
  const float* bq = (const float*)d_in[5];
  const float* Wk = (const float*)d_in[6];
  const float* bk = (const float*)d_in[7];
  const float* Wv = (const float*)d_in[8];
  const float* bv = (const float*)d_in[9];
  const float* Wo = (const float*)d_in[10];
  const float* bo = (const float*)d_in[11];

  const size_t nTok = (size_t)B_ * T_ * D_;   // == B*S*D
  char* ws = (char*)d_ws;
  unsigned short* Qp = (unsigned short*)ws;  ws += nTok * 2;
  unsigned short* Kp = (unsigned short*)ws;  ws += nTok * 2;
  unsigned short* Vp = (unsigned short*)ws;  ws += nTok * 2;
  float* attn = (float*)ws;                  // B*T*D f32

  const dim3 ggrid(B_ * T_ / 128, D_ / 64);
  gemm_bias<true><<<ggrid, 256, 0, stream>>>(q, Wq, bq, (void*)Qp);
  gemm_bias<true><<<ggrid, 256, 0, stream>>>(k, Wk, bk, (void*)Kp);
  gemm_bias<true><<<ggrid, 256, 0, stream>>>(v, Wv, bv, (void*)Vp);

  attention<<<dim3(T_ / 64, H_, B_), 128, 0, stream>>>(Qp, Kp, Vp, mask, attn);

  gemm_bias<false><<<ggrid, 256, 0, stream>>>(attn, Wo, bo, d_out);
}
